// MistralLLM_45878840656340
// MI455X (gfx1250) — matrix operand order repaired and checked by reading, not compiled, not run on hardware
//
#include <hip/hip_runtime.h>
#include <hip/hip_bf16.h>
#include <math.h>

// ---------------------------------------------------------------------------
// MI455X (gfx1250, wave32) implementation of the 2-layer LoRA-VLM forward.
// All big matmuls use v_wmma_f32_16x16x32_f16 with f32 accumulation.
// GEMM is software-pipelined: global->regs (tile k+1) overlaps WMMA (tile k),
// with double-buffered LDS and one barrier per K-step.
// ---------------------------------------------------------------------------

typedef __attribute__((ext_vector_type(16))) _Float16 v16h;
typedef __attribute__((ext_vector_type(8)))  _Float16 v8h;
typedef __attribute__((ext_vector_type(8)))  float    v8f;

constexpr int Dd    = 4096;
constexpr int VHc   = 1024;
constexpr int DFFc  = 14336;
constexpr int NLc   = 2;
constexpr int NHc   = 32;
constexpr int NKVc  = 8;
constexpr int HDc   = 128;
constexpr int Rc    = 8;
constexpr int Bc    = 2;
constexpr int NIMGc = 257;
constexpr int Tc    = 511;
constexpr int Sc    = NIMGc + Tc;   // 768
constexpr int BSc   = Bc * Sc;      // 1536
constexpr int KVD   = NKVc * HDc;   // 1024
constexpr float LORA_SCALE = 4.0f;

enum { EPI_STORE = 0, EPI_BIAS = 1, EPI_ADD = 2, EPI_SILU = 3, EPI_MUL = 4 };

static __device__ __forceinline__ v16h cat16(v8h lo, v8h hi) {
  return __builtin_shufflevector(lo, hi, 0,1,2,3,4,5,6,7,8,9,10,11,12,13,14,15);
}
static __device__ __forceinline__ v8f wmma_f16(v16h a, v16h b, v8f c) {
  // (neg_a, A, neg_b, B, c_mod, C, reuse_a, reuse_b)
  return __builtin_amdgcn_wmma_f32_16x16x32_f16(false, a, false, b, (short)0, c, false, false);
}

// ---------------------------------------------------------------------------
// Generic GEMM: C[M,N] = epi( A[M,K] * W[N,K]^T  (+ LoRA rank-8 correction) )
// Tile: 256 (M) x 64 (N), KT=32. 8 waves; wave w owns rows [w*32, w*32+32),
// each wave computes 2 (M) x 4 (N) WMMA 16x16 tiles.
// LDS rows padded to 40 halfs (80B = 20 banks) -> conflict-free b128 reads.
// Double-buffered LDS; next tile's globals staged in registers during WMMA.
// ---------------------------------------------------------------------------
__global__ __launch_bounds__(256) void gemm_f16w(
    const float* __restrict__ A, const float* __restrict__ W,
    float* __restrict__ C, const float* __restrict__ Cin,
    const float* __restrict__ Tl, const float* __restrict__ Bm,
    int M, int N, int K, int epi)
{
  __shared__ __align__(16) _Float16 Ash[2][256][40];
  __shared__ __align__(16) _Float16 Bsh[2][64][40];

  const int tid  = threadIdx.x;
  const int n0   = blockIdx.x * 64;
  const int m0   = blockIdx.y * 256;
  const int wid  = tid >> 5;
  const int lane = tid & 31;
  const int lh   = lane & 15;
  const bool hiL = lane >= 16;

  v8f acc[2][4] = {};

  const int ar = tid >> 3;        // 0..31  (row within a 32-row pass)
  const int ac = (tid & 7) * 4;   // 0,4,..,28 (half/float column)

  // Per-thread global row pointers (row clamp is K-invariant -> hoisted)
  const float* arow[8];
#pragma unroll
  for (int p = 0; p < 8; ++p) {
    int grow = m0 + p * 32 + ar; if (grow >= M) grow = M - 1;
    arow[p] = A + (size_t)grow * K + ac;
  }
  const float* brow[2];
#pragma unroll
  for (int p = 0; p < 2; ++p) brow[p] = W + (size_t)(n0 + p * 32 + ar) * K + ac;

  float4 ra[8], rb[2];

  auto load_regs = [&](int k0) {
#pragma unroll
    for (int p = 0; p < 8; ++p) ra[p] = *(const float4*)(arow[p] + k0);
#pragma unroll
    for (int p = 0; p < 2; ++p) rb[p] = *(const float4*)(brow[p] + k0);
  };
  auto store_lds = [&](int buf) {
#pragma unroll
    for (int p = 0; p < 8; ++p) {
      union { _Float16 h[4]; uint2 u; } pk;
      pk.h[0] = (_Float16)ra[p].x; pk.h[1] = (_Float16)ra[p].y;
      pk.h[2] = (_Float16)ra[p].z; pk.h[3] = (_Float16)ra[p].w;
      *(uint2*)&Ash[buf][p * 32 + ar][ac] = pk.u;
    }
#pragma unroll
    for (int p = 0; p < 2; ++p) {
      union { _Float16 h[4]; uint2 u; } pk;
      pk.h[0] = (_Float16)rb[p].x; pk.h[1] = (_Float16)rb[p].y;
      pk.h[2] = (_Float16)rb[p].z; pk.h[3] = (_Float16)rb[p].w;
      *(uint2*)&Bsh[buf][p * 32 + ar][ac] = pk.u;
    }
  };

  // Prologue: stage tile 0
  load_regs(0);
  store_lds(0);
  __syncthreads();

  int buf = 0;
  for (int k0 = 0; k0 < K; k0 += 32) {
    const bool more = (k0 + 32 < K);
    if (more) {
      load_regs(k0 + 32);           // issue next tile's loads before compute
      if (k0 + 96 < K) {            // stream weights two tiles ahead into L2
        __builtin_prefetch(brow[0] + k0 + 64, 0, 0);
        __builtin_prefetch(brow[1] + k0 + 64, 0, 0);
      }
    }

    // ---- fragments & WMMA on current LDS buffer ----
    const int koffA = hiL ? 16 : 0;  // A: lanes 0-15 hold K 0..15, hi K 16..31
    const int r0 = wid * 32 + lh;
    v16h a0 = cat16(*(const v8h*)&Ash[buf][r0     ][koffA], *(const v8h*)&Ash[buf][r0     ][8 + koffA]);
    v16h a1 = cat16(*(const v8h*)&Ash[buf][r0 + 16][koffA], *(const v8h*)&Ash[buf][r0 + 16][8 + koffA]);
    const int koffB = hiL ? 16 : 0;  // B: lanes 0-15 hold K 0..15, hi K 16..31
#pragma unroll
    for (int tn = 0; tn < 4; ++tn) {
      int cc = tn * 16 + lh;
      v16h bb = cat16(*(const v8h*)&Bsh[buf][cc][koffB], *(const v8h*)&Bsh[buf][cc][koffB + 8]);
      acc[0][tn] = wmma_f16(a0, bb, acc[0][tn]);
      acc[1][tn] = wmma_f16(a1, bb, acc[1][tn]);
    }

    if (more) {
      store_lds(buf ^ 1);           // convert + stage into the other buffer
      __syncthreads();              // single barrier per K-step
      buf ^= 1;
    }
  }

  // ---- epilogue: LoRA correction + fused op ----
#pragma unroll
  for (int tn = 0; tn < 4; ++tn) {
    const int colg = n0 + tn * 16 + lh;
    float lb[8];
    if (Tl) {
      const float4 b0v = *(const float4*)(Bm + (size_t)colg * 8);
      const float4 b1v = *(const float4*)(Bm + (size_t)colg * 8 + 4);
      lb[0]=b0v.x; lb[1]=b0v.y; lb[2]=b0v.z; lb[3]=b0v.w;
      lb[4]=b1v.x; lb[5]=b1v.y; lb[6]=b1v.z; lb[7]=b1v.w;
    }
#pragma unroll
    for (int tm = 0; tm < 2; ++tm) {
      v8f c = acc[tm][tn];
      const int rowb = m0 + wid * 32 + tm * 16 + (hiL ? 8 : 0);
#pragma unroll
      for (int j = 0; j < 8; ++j) {
        int row = rowb + j;
        if (row < M) {
          float r = c[j];
          if (Tl) {
            const float* tr = Tl + (size_t)row * 8;
            float s = tr[0]*lb[0] + tr[1]*lb[1] + tr[2]*lb[2] + tr[3]*lb[3]
                    + tr[4]*lb[4] + tr[5]*lb[5] + tr[6]*lb[6] + tr[7]*lb[7];
            r += LORA_SCALE * s;
          }
          size_t idx = (size_t)row * N + colg;
          switch (epi) {
            case EPI_STORE: C[idx] = r;                              break;
            case EPI_BIAS:  C[idx] = r + Cin[colg];                  break;
            case EPI_ADD:   C[idx] = Cin[idx] + r;                   break;
            case EPI_SILU:  C[idx] = r / (1.0f + __expf(-r));        break;
            case EPI_MUL:   C[idx] = Cin[idx] * r;                   break;
          }
        }
      }
    }
  }
}

// ---------------------------------------------------------------------------
// RMSNorm (one block per row)
// ---------------------------------------------------------------------------
__global__ __launch_bounds__(256) void rmsnorm_k(const float* __restrict__ H,
                                                 const float* __restrict__ w,
                                                 float* __restrict__ X, int Kd)
{
  __shared__ float red[256];
  const int m = blockIdx.x;
  const float* hr = H + (size_t)m * Kd;
  float s = 0.0f;
  for (int k = threadIdx.x; k < Kd; k += 256) { float v = hr[k]; s += v * v; }
  red[threadIdx.x] = s; __syncthreads();
  for (int o = 128; o > 0; o >>= 1) {
    if (threadIdx.x < o) red[threadIdx.x] += red[threadIdx.x + o];
    __syncthreads();
  }
  const float inv = rsqrtf(red[0] / (float)Kd + 1e-5f);
  float* xr = X + (size_t)m * Kd;
  for (int k = threadIdx.x; k < Kd; k += 256) xr[k] = hr[k] * inv * w[k];
}

// ---------------------------------------------------------------------------
// t[m, r] = dot(x[m,:], A[r,:])  — one block per row, wave w handles rank r=w
// ---------------------------------------------------------------------------
__global__ __launch_bounds__(256) void lora_t_k(const float* __restrict__ X,
                                                const float* __restrict__ Am,
                                                float* __restrict__ T, int Kd)
{
  const int m = blockIdx.x;
  const int r = threadIdx.x >> 5, lane = threadIdx.x & 31;
  const float* x = X + (size_t)m * Kd;
  const float* a = Am + (size_t)r * Kd;
  float s = 0.0f;
  for (int k = lane; k < Kd; k += 32) s += x[k] * a[k];
  for (int o = 16; o > 0; o >>= 1) s += __shfl_xor(s, o, 32);
  if (lane == 0) T[(size_t)m * 8 + r] = s;
}

// ---------------------------------------------------------------------------
// Token embedding lookup into the text slots of h
// ---------------------------------------------------------------------------
__global__ __launch_bounds__(256) void embed_k(const int* __restrict__ ids,
                                               const float* __restrict__ E,
                                               float* __restrict__ Hs)
{
  const int bt = blockIdx.x;
  const int b = bt / Tc, t = bt % Tc;
  const float* src = E + (size_t)ids[bt] * Dd;
  float* dst = Hs + (size_t)(b * Sc + NIMGc + t) * Dd;
  for (int k = threadIdx.x; k < Dd; k += 256) dst[k] = src[k];
}

// ---------------------------------------------------------------------------
// RoPE + f16 packing.  Q: (BS, 4096)f32 -> [b][h][s][hd] f16 (rope applied)
// ---------------------------------------------------------------------------
__global__ __launch_bounds__(256) void rope_q_k(const float* __restrict__ Q,
                                                _Float16* __restrict__ Qh)
{
  const int row = blockIdx.x;                 // b*S + s
  const int b = row / Sc, s = row % Sc;
  const float* q = Q + (size_t)row * Dd;
  for (int e = threadIdx.x; e < Dd; e += 256) {
    int hh = e >> 7, d = e & 127, j = d & 63;
    float ang = (float)s * __powf(10000.0f, -(float)j * (1.0f / 64.0f));
    float sn, c; __sincosf(ang, &sn, &c);
    float xv = q[e];
    float xr = (d < 64) ? -q[hh * 128 + d + 64] : q[hh * 128 + d - 64];
    Qh[((size_t)(b * NHc + hh) * Sc + s) * HDc + d] = (_Float16)(xv * c + xr * sn);
  }
}

__global__ __launch_bounds__(256) void rope_kk_k(const float* __restrict__ Kb,
                                                 _Float16* __restrict__ Kh)
{
  const int row = blockIdx.x;
  const int b = row / Sc, s = row % Sc;
  const float* kq = Kb + (size_t)row * KVD;
  for (int e = threadIdx.x; e < KVD; e += 256) {
    int hh = e >> 7, d = e & 127, j = d & 63;
    float ang = (float)s * __powf(10000.0f, -(float)j * (1.0f / 64.0f));
    float sn, c; __sincosf(ang, &sn, &c);
    float xv = kq[e];
    float xr = (d < 64) ? -kq[hh * 128 + d + 64] : kq[hh * 128 + d - 64];
    Kh[((size_t)(b * NKVc + hh) * Sc + s) * HDc + d] = (_Float16)(xv * c + xr * sn);
  }
}

// V packed transposed: Vt[b][kvh][hd][s] so P@V B-fragments are contiguous
__global__ __launch_bounds__(256) void packvt_k(const float* __restrict__ V,
                                                _Float16* __restrict__ Vt)
{
  const int row = blockIdx.x;
  const int b = row / Sc, s = row % Sc;
  const float* v = V + (size_t)row * KVD;
  for (int e = threadIdx.x; e < KVD; e += 256) {
    int hh = e >> 7, d = e & 127;
    Vt[((size_t)(b * NKVc + hh) * HDc + d) * Sc + s] = (_Float16)v[e];
  }
}

// ---------------------------------------------------------------------------
// Attention: one block = 16 query rows of one (b, h); full 768-key panel in
// LDS (f16).  QK^T and P@V both via WMMA.  Output un-normalized P@V divided
// by the softmax row-sum in the epilogue.
// ---------------------------------------------------------------------------
constexpr int SW = 776;   // padded panel stride (halfs)

__global__ __launch_bounds__(256) void attn_k(const _Float16* __restrict__ Qh,
                                              const _Float16* __restrict__ Kh,
                                              const _Float16* __restrict__ Vt,
                                              const int* __restrict__ amask,
                                              float* __restrict__ Ob)
{
  __shared__ __align__(16) _Float16 P[16 * SW];   // 24.8 KB score/prob panel
  __shared__ float rowsum[16];

  const int bh = blockIdx.x, qt = blockIdx.y;
  const int b = bh / NHc, h = bh % NHc, kvh = h >> 2;   // GQA: 4 q-heads per kv-head
  const int q0 = qt * 16;
  const int tid = threadIdx.x, wid = tid >> 5, lane = tid & 31, lh = lane & 15;
  const bool hiL = lane >= 16;

  const _Float16* Qb = Qh + (size_t)(b * NHc  + h  ) * Sc * HDc;
  const _Float16* Kb = Kh + (size_t)(b * NKVc + kvh) * Sc * HDc;
  const _Float16* Vb = Vt + (size_t)(b * NKVc + kvh) * HDc * Sc;

  // Q fragments (re-used across all key tiles)
  v16h qa[4];
  {
    const _Float16* qr = Qb + (size_t)(q0 + lh) * HDc;
#pragma unroll
    for (int kk = 0; kk < 4; ++kk) {
      int koff = kk * 32 + (hiL ? 16 : 0);
      qa[kk] = cat16(*(const v8h*)(qr + koff), *(const v8h*)(qr + koff + 8));
    }
  }

  // Phase 1: scores -> f16 panel (masked, scaled). Wave w owns 6 key tiles.
  for (int kt = wid * 6; kt < wid * 6 + 6; ++kt) {
    const int kcol = kt * 16 + lh;
    const _Float16* kr = Kb + (size_t)kcol * HDc;
    v8f c = {};
#pragma unroll
    for (int kk = 0; kk < 4; ++kk) {
      int koff = kk * 32 + (hiL ? 16 : 0);
      v16h bb = cat16(*(const v8h*)(kr + koff), *(const v8h*)(kr + koff + 8));
      c = wmma_f16(qa[kk], bb, c);
    }
    const bool mk = amask[b * Sc + kcol] > 0;
#pragma unroll
    for (int j = 0; j < 8; ++j) {
      int rl = j + (hiL ? 8 : 0);
      int rowg = q0 + rl;
      float sc = c[j] * 0.08838834764831845f;        // 1/sqrt(128)
      bool ok = mk && (kcol <= rowg);                // causal + attention_mask
      P[rl * SW + kcol] = (_Float16)(ok ? sc : -30000.0f);
    }
  }
  __syncthreads();

  // Phase 2: softmax (16-lane group per row); store exp(s-max), keep row sums
  {
    const int g = tid >> 4, li = tid & 15;
    float mx = -1.0e30f;
    for (int cix = li; cix < Sc; cix += 16) mx = fmaxf(mx, (float)P[g * SW + cix]);
#pragma unroll
    for (int o = 8; o > 0; o >>= 1) mx = fmaxf(mx, __shfl_xor(mx, o, 32));
    float sm = 0.0f;
    for (int cix = li; cix < Sc; cix += 16) {
      float e = __expf((float)P[g * SW + cix] - mx);
      P[g * SW + cix] = (_Float16)e;
      sm += e;
    }
#pragma unroll
    for (int o = 8; o > 0; o >>= 1) sm += __shfl_xor(sm, o, 32);
    if (li == 0) rowsum[g] = sm;
  }
  __syncthreads();

  // Phase 3: (P @ V) / rowsum.  Wave w owns hd columns [w*16, w*16+16).
  {
    const int ncol = wid * 16 + lh;
    const _Float16* vr = Vb + (size_t)ncol * Sc;
    v8f c = {};
    for (int kk = 0; kk < Sc / 32; ++kk) {
      int k0 = kk * 32;
      int ka = k0 + (hiL ? 16 : 0);
      v16h aa = cat16(*(const v8h*)&P[lh * SW + ka], *(const v8h*)&P[lh * SW + ka + 8]);
      int kb2 = k0 + (hiL ? 16 : 0);
      v16h bb = cat16(*(const v8h*)(vr + kb2), *(const v8h*)(vr + kb2 + 8));
      c = wmma_f16(aa, bb, c);
    }
#pragma unroll
    for (int j = 0; j < 8; ++j) {
      int rl = j + (hiL ? 8 : 0);
      float o = c[j] / rowsum[rl];
      Ob[(size_t)(b * Sc + q0 + rl) * Dd + h * HDc + ncol] = o;
    }
  }
}

// ---------------------------------------------------------------------------
// Host orchestration
// ---------------------------------------------------------------------------
extern "C" void kernel_launch(void* const* d_in, const int* in_sizes, int n_in,
                              void* d_out, int out_size, void* d_ws, size_t ws_size,
                              hipStream_t stream)
{
  (void)in_sizes; (void)n_in; (void)out_size; (void)ws_size;

  const float* image_embeds = (const float*)d_in[0];
  const int*   input_ids    = (const int*)  d_in[1];
  const int*   amask        = (const int*)  d_in[2];
  const float* proj_W = (const float*)d_in[3];
  const float* proj_b = (const float*)d_in[4];
  const float* embedW = (const float*)d_in[5];
  const float* ln1  = (const float*)d_in[6];
  const float* ln2  = (const float*)d_in[7];
  const float* ln_f = (const float*)d_in[8];
  const float* Wq = (const float*)d_in[9];  const float* Aq = (const float*)d_in[10]; const float* Bq = (const float*)d_in[11];
  const float* Wk = (const float*)d_in[12]; const float* Ak = (const float*)d_in[13]; const float* Bk = (const float*)d_in[14];
  const float* Wv = (const float*)d_in[15]; const float* Av = (const float*)d_in[16]; const float* Bv = (const float*)d_in[17];
  const float* Wo = (const float*)d_in[18]; const float* Ao = (const float*)d_in[19]; const float* Bo = (const float*)d_in[20];
  const float* Wg = (const float*)d_in[21]; const float* Ag = (const float*)d_in[22]; const float* Bg = (const float*)d_in[23];
  const float* Wu = (const float*)d_in[24]; const float* Au = (const float*)d_in[25]; const float* Bu = (const float*)d_in[26];
  const float* Wd = (const float*)d_in[27]; const float* Ad = (const float*)d_in[28]; const float* Bd = (const float*)d_in[29];

  char* wsb = (char*)d_ws;
  size_t off = 0;
  auto alloc = [&](size_t bytes) -> void* {
    void* p = (void*)(wsb + off);
    off += (bytes + 255) & ~(size_t)255;
    return p;
  };
  float* h  = (float*)alloc((size_t)BSc * Dd  * 4);   // residual stream
  float* x  = (float*)alloc((size_t)BSc * Dd  * 4);   // normed activations
  float* qb = (float*)alloc((size_t)BSc * Dd  * 4);   // q f32; reused as attention output
  float* kb = (float*)alloc((size_t)BSc * KVD * 4);
  float* vb = (float*)alloc((size_t)BSc * KVD * 4);
  float* G  = (float*)alloc((size_t)BSc * DFFc * 4);  // gate -> silu(g)*u in place
  float* tb = (float*)alloc((size_t)BSc * 8 * 4);     // LoRA t = x A^T
  _Float16* qh = (_Float16*)alloc((size_t)BSc * Dd  * 2);
  _Float16* kh = (_Float16*)alloc((size_t)BSc * KVD * 2);
  _Float16* vt = (_Float16*)alloc((size_t)BSc * KVD * 2);

  // Projector (per batch: output rows not contiguous across batches) + bias
  for (int b = 0; b < Bc; ++b) {
    gemm_f16w<<<dim3(Dd / 64, (NIMGc + 255) / 256), 256, 0, stream>>>(
        image_embeds + (size_t)b * NIMGc * VHc, proj_W,
        h + (size_t)b * Sc * Dd, proj_b, nullptr, nullptr,
        NIMGc, Dd, VHc, EPI_BIAS);
  }
  embed_k<<<Bc * Tc, 256, 0, stream>>>(input_ids, embedW, h);

  const dim3 gD  (Dd   / 64, BSc / 256);
  const dim3 gKV (KVD  / 64, BSc / 256);
  const dim3 gFF (DFFc / 64, BSc / 256);

  for (int L = 0; L < NLc; ++L) {
    const float* wq = Wq + (size_t)L * Dd * Dd;    const float* aq = Aq + (size_t)L * Rc * Dd;   const float* bq = Bq + (size_t)L * Dd * Rc;
    const float* wk = Wk + (size_t)L * KVD * Dd;   const float* ak = Ak + (size_t)L * Rc * Dd;   const float* bk = Bk + (size_t)L * KVD * Rc;
    const float* wv = Wv + (size_t)L * KVD * Dd;   const float* av = Av + (size_t)L * Rc * Dd;   const float* bv = Bv + (size_t)L * KVD * Rc;
    const float* wo = Wo + (size_t)L * Dd * Dd;    const float* ao = Ao + (size_t)L * Rc * Dd;   const float* bo = Bo + (size_t)L * Dd * Rc;
    const float* wg = Wg + (size_t)L * DFFc * Dd;  const float* ag = Ag + (size_t)L * Rc * Dd;   const float* bg = Bg + (size_t)L * DFFc * Rc;
    const float* wu = Wu + (size_t)L * DFFc * Dd;  const float* au = Au + (size_t)L * Rc * Dd;   const float* bu = Bu + (size_t)L * DFFc * Rc;
    const float* wd = Wd + (size_t)L * Dd * DFFc;  const float* ad = Ad + (size_t)L * Rc * DFFc; const float* bd = Bd + (size_t)L * Dd * Rc;

    // --- attention block ---
    rmsnorm_k<<<BSc, 256, 0, stream>>>(h, ln1 + (size_t)L * Dd, x, Dd);

    lora_t_k<<<BSc, 256, 0, stream>>>(x, aq, tb, Dd);
    gemm_f16w<<<gD, 256, 0, stream>>>(x, wq, qb, nullptr, tb, bq, BSc, Dd, Dd, EPI_STORE);
    rope_q_k<<<BSc, 256, 0, stream>>>(qb, qh);

    lora_t_k<<<BSc, 256, 0, stream>>>(x, ak, tb, Dd);
    gemm_f16w<<<gKV, 256, 0, stream>>>(x, wk, kb, nullptr, tb, bk, BSc, KVD, Dd, EPI_STORE);
    rope_kk_k<<<BSc, 256, 0, stream>>>(kb, kh);

    lora_t_k<<<BSc, 256, 0, stream>>>(x, av, tb, Dd);
    gemm_f16w<<<gKV, 256, 0, stream>>>(x, wv, vb, nullptr, tb, bv, BSc, KVD, Dd, EPI_STORE);
    packvt_k<<<BSc, 256, 0, stream>>>(vb, vt);

    attn_k<<<dim3(Bc * NHc, Sc / 16), 256, 0, stream>>>(qh, kh, vt, amask, qb);

    lora_t_k<<<BSc, 256, 0, stream>>>(qb, ao, tb, Dd);
    gemm_f16w<<<gD, 256, 0, stream>>>(qb, wo, h, h, tb, bo, BSc, Dd, Dd, EPI_ADD);

    // --- MLP block ---
    rmsnorm_k<<<BSc, 256, 0, stream>>>(h, ln2 + (size_t)L * Dd, x, Dd);

    lora_t_k<<<BSc, 256, 0, stream>>>(x, ag, tb, Dd);
    gemm_f16w<<<gFF, 256, 0, stream>>>(x, wg, G, nullptr, tb, bg, BSc, DFFc, Dd, EPI_SILU);

    lora_t_k<<<BSc, 256, 0, stream>>>(x, au, tb, Dd);
    gemm_f16w<<<gFF, 256, 0, stream>>>(x, wu, G, G, tb, bu, BSc, DFFc, Dd, EPI_MUL);

    lora_t_k<<<BSc, 256, 0, stream>>>(G, ad, tb, DFFc);
    gemm_f16w<<<gD, 256, 0, stream>>>(G, wd, h, h, tb, bd, BSc, Dd, DFFc, EPI_ADD);
  }

  rmsnorm_k<<<BSc, 256, 0, stream>>>(h, ln_f, (float*)d_out, Dd);
}
